// SelfAttention_549755814602
// MI455X (gfx1250) — compile-verified, matrix operand order checked by reading, not run on hardware
//
#include <hip/hip_runtime.h>
#include <hip/hip_bf16.h>

#define EMBED 512
#define HEADS 8
#define HD    64
#define BATCH 2
#define SEQ   4096

typedef __attribute__((ext_vector_type(16))) _Float16 v16h;
typedef __attribute__((ext_vector_type(8)))  _Float16 v8h;
typedef __attribute__((ext_vector_type(8)))  float    v8f;
typedef unsigned int u32x4 __attribute__((ext_vector_type(4)));
typedef int          i32x4 __attribute__((ext_vector_type(4)));
typedef int          i32x8 __attribute__((ext_vector_type(8)));

static __device__ inline v8f wmma16(v16h a, v16h b, v8f c) {
  // D = A(16x32 f16) * B(32x16 f16) + C(16x16 f32)
  return __builtin_amdgcn_wmma_f32_16x16x32_f16(false, a, false, b,
                                                (short)0, c, false, false);
}

// Load 16 halves for an A/B operand lane: chunk0 = p[base..base+7],
// chunk1 = p[base+16..base+23] (base = 0 for lanes 0-15, 8 for lanes 16-31),
// matching the ISA 16-bit 16x32 operand VGPR striping.
static __device__ inline v16h load_a16(const _Float16* p, int base) {
  const v8h c0 = *(const v8h*)(p + base);
  const v8h c1 = *(const v8h*)(p + base + 16);
  v16h r;
#pragma unroll
  for (int i = 0; i < 8; ++i) { r[i] = c0[i]; r[8 + i] = c1[i]; }
  return r;
}

static __device__ inline v16h load_a32(const float* p, int base) {
  v16h r;
#pragma unroll
  for (int i = 0; i < 8; ++i) r[i] = (_Float16)p[base + i];
#pragma unroll
  for (int i = 0; i < 8; ++i) r[8 + i] = (_Float16)p[base + 16 + i];
  return r;
}

// ---------------------------------------------------------------------------
// Tensor Data Mover: issue a 2D (or 1D with td1=tl1=1) tile load into LDS.
// All dims/strides in 8-byte units (D#.data_size = 3). Descriptor layout per
// CDNA5 ISA ch.8: group0 = {count, lds_addr, global_addr, type=2},
// group1 = {data_size, tensor_dim0/1, tile_dim0/1, tensor_dim0_stride}.
// ---------------------------------------------------------------------------
static __device__ inline void tdm_load_to_lds(unsigned lds_off,
                                              unsigned long long ga,
                                              unsigned td0, unsigned td1,
                                              unsigned tl0, unsigned tl1,
                                              unsigned stride0) {
  u32x4 g0;
  g0[0] = 1u;                                    // count = 1 valid descriptor
  g0[1] = lds_off;                               // lds_addr (bytes)
  g0[2] = (unsigned)ga;                          // global_addr[31:0]
  g0[3] = (unsigned)((ga >> 32) & 0x01FFFFFFu)   // global_addr[56:32]
          | 0x80000000u;                         // type = 2 ("image")
  i32x8 g1;
  g1[0] = (int)(3u << 16);                       // data_size = 8 bytes
  g1[1] = (int)(td0 << 16);                      // tensor_dim0[15:0]
  g1[2] = (int)((td0 >> 16) | (td1 << 16));      // tensor_dim0[31:16] | dim1
  g1[3] = (int)((td1 >> 16) | (tl0 << 16));      // tensor_dim1[31:16] | tile0
  g1[4] = (int)(tl1 & 0xFFFFu);                  // tile_dim1 (tile_dim2 = 0)
  g1[5] = (int)stride0;                          // tensor_dim0_stride[31:0]
  g1[6] = 0;                                     // stride0 hi | stride1 lo
  g1[7] = 0;
  i32x4 z4 = {0, 0, 0, 0};
#if defined(__clang_major__) && __clang_major__ >= 23
  i32x8 z8 = {0, 0, 0, 0, 0, 0, 0, 0};
  __builtin_amdgcn_tensor_load_to_lds(g0, g1, z4, z4, z8, 0);
#else
  __builtin_amdgcn_tensor_load_to_lds(g0, g1, z4, z4, 0);
#endif
}

static __device__ inline unsigned lds_off32(const void* p) {
  return (unsigned)(unsigned long long)(uintptr_t)p;
}

// ---------------------------------------------------------------------------
// Kernel 1: per-head QKV projections (x @ W^T), f32 -> f16, softmax scale
// folded into Q. Q,K stored [B*H, S, 64]; V stored transposed [B*H, 64, S].
// ---------------------------------------------------------------------------
__global__ void qkv_project_kernel(const float* __restrict__ vin,
                                   const float* __restrict__ kin,
                                   const float* __restrict__ qin,
                                   const float* __restrict__ Wv,
                                   const float* __restrict__ Wk,
                                   const float* __restrict__ Wq,
                                   _Float16* __restrict__ q_ws,
                                   _Float16* __restrict__ k_ws,
                                   _Float16* __restrict__ vT_ws) {
  const int lane = threadIdx.x & 31;
  const int wave = threadIdx.x >> 5;
  const int bh = blockIdx.x;
  const int b  = bh / HEADS, h = bh % HEADS;
  const int s0 = blockIdx.y * 64 + wave * 16;
  const int base = (lane < 16) ? 0 : 8;
  const int col  = lane & 15;
  const int hi8  = (lane < 16) ? 0 : 8;
  const float SCALE = 0.044194173824159216f;   // 1/sqrt(EMBED)

  const float* xin[3] = {qin, kin, vin};
  const float* W[3]   = {Wq, Wk, Wv};
  const int row = s0 + col;

#pragma unroll
  for (int m = 0; m < 3; ++m) {
    const float* xrow = xin[m] + ((size_t)b * SEQ + row) * EMBED + h * HD;
    v16h a0 = load_a32(xrow, base);
    v16h a1 = load_a32(xrow + 32, base);
#pragma unroll
    for (int nt = 0; nt < 4; ++nt) {
      const int n0 = nt * 16;
      const float* wrow = W[m] + (size_t)(n0 + col) * HD;
      v16h b0 = load_a32(wrow, base);
      v16h b1 = load_a32(wrow + 32, base);
      v8f acc = {};
      acc = wmma16(a0, b0, acc);
      acc = wmma16(a1, b1, acc);
      if (m == 0) {
#pragma unroll
        for (int j = 0; j < 8; ++j)
          q_ws[((size_t)bh * SEQ + (s0 + j + hi8)) * HD + n0 + col] =
              (_Float16)(acc[j] * SCALE);
      } else if (m == 1) {
#pragma unroll
        for (int j = 0; j < 8; ++j)
          k_ws[((size_t)bh * SEQ + (s0 + j + hi8)) * HD + n0 + col] =
              (_Float16)acc[j];
      } else {
#pragma unroll
        for (int j = 0; j < 8; ++j)
          vT_ws[((size_t)bh * HD + n0 + col) * SEQ + (s0 + j + hi8)] =
              (_Float16)acc[j];
      }
    }
  }
}

// ---------------------------------------------------------------------------
// Kernel 2: flash attention. Block = 4 waves sharing one (b,h); K/V tiles are
// staged into LDS by the Tensor Data Mover (double-buffered, TENSORcnt) and
// consumed by all waves. One wave owns 16 query rows; O + (m_i,l_i) live in
// registers; P transposed C-layout -> A-layout through a per-wave LDS tile.
// ---------------------------------------------------------------------------
__global__ void flash_attn_kernel(const _Float16* __restrict__ q_ws,
                                  const _Float16* __restrict__ k_ws,
                                  const _Float16* __restrict__ vT_ws,
                                  _Float16* __restrict__ attn_ws) {
  __shared__ __align__(16) _Float16 kbuf[2][32 * HD];   // [token][d]   4KB x2
  __shared__ __align__(16) _Float16 vbuf[2][HD * 32];   // [d][token]   4KB x2
  __shared__ __align__(16) _Float16 pbuf[4][16 * 32];   // P transpose  1KB x4

  const int lane = threadIdx.x & 31;
  const int wave = threadIdx.x >> 5;
  const int bh = blockIdx.x;
  const int b  = bh / HEADS, h = bh % HEADS;
  const int s0 = blockIdx.y * 64 + wave * 16;
  const int base = (lane < 16) ? 0 : 8;
  const int col  = lane & 15;
  const int hi8  = (lane < 16) ? 0 : 8;

  const _Float16* qrow = q_ws + ((size_t)bh * SEQ + (s0 + col)) * HD;
  v16h qa0 = load_a16(qrow, base);
  v16h qa1 = load_a16(qrow + 32, base);

  float mi[8], li[8];
  v8f acc[4];
#pragma unroll
  for (int j = 0; j < 8; ++j) { mi[j] = -1e30f; li[j] = 0.0f; }
#pragma unroll
  for (int t = 0; t < 4; ++t) acc[t] = (v8f){};

  const _Float16* kbase = k_ws + (size_t)bh * SEQ * HD;
  const _Float16* vbase = vT_ws + (size_t)bh * HD * SEQ;
  _Float16* pb = pbuf[wave];

  const unsigned koff[2] = {lds_off32(&kbuf[0][0]), lds_off32(&kbuf[1][0])};
  const unsigned voff[2] = {lds_off32(&vbuf[0][0]), lds_off32(&vbuf[1][0])};

  const int NT = SEQ / 32;   // 128 key tiles

  // Prologue: TDM-load tile 0 into buffer 0 (wave 0 drives the DMA engine).
  if (wave == 0) {
    // K tile: 32 rows x 64 f16 contiguous = 512 x 8B, 1D.
    tdm_load_to_lds(koff[0], (unsigned long long)(uintptr_t)kbase,
                    512u, 1u, 512u, 1u, 512u);
    // V tile: 64 rows x 64B from vT (row stride SEQ*2B = 1024 x 8B), 2D.
    tdm_load_to_lds(voff[0], (unsigned long long)(uintptr_t)vbase,
                    1024u, 64u, 8u, 64u, 1024u);
  }

  for (int i = 0; i < NT; ++i) {
    const int kb = i * 32;
    const int cur = i & 1;
    if (wave == 0) {
      if (i + 1 < NT) {   // overlap: issue next tile into the other buffer
        const int kb2 = kb + 32;
        tdm_load_to_lds(koff[cur ^ 1],
                        (unsigned long long)(uintptr_t)(kbase + (size_t)kb2 * HD),
                        512u, 1u, 512u, 1u, 512u);
        tdm_load_to_lds(voff[cur ^ 1],
                        (unsigned long long)(uintptr_t)(vbase + kb2),
                        1024u, 64u, 8u, 64u, 1024u);
        __builtin_amdgcn_s_wait_tensorcnt((short)2);  // tile i arrived
      } else {
        __builtin_amdgcn_s_wait_tensorcnt((short)0);
      }
    }
    __syncthreads();   // publish tile i to all waves

    // ---- scores S = Q @ K^T (pre-scaled Q), two 16-key subtiles ----
    const _Float16* krow0 = &kbuf[cur][(size_t)col * HD];
    const _Float16* krow1 = &kbuf[cur][(size_t)(16 + col) * HD];
    v8f s0a = {}, s1a = {};
    {
      v16h b0 = load_a16(krow0, base);
      v16h b1 = load_a16(krow0 + 32, base);
      s0a = wmma16(qa0, b0, s0a);
      s0a = wmma16(qa1, b1, s0a);
      v16h c0 = load_a16(krow1, base);
      v16h c1 = load_a16(krow1 + 32, base);
      s1a = wmma16(qa0, c0, s1a);
      s1a = wmma16(qa1, c1, s1a);
    }

    // ---- online softmax: half-wave reductions match C-layout rows ----
    float p0[8], p1[8];
#pragma unroll
    for (int j = 0; j < 8; ++j) {
      float v = fmaxf(s0a[j], s1a[j]);
      v = fmaxf(v, __shfl_xor(v, 1, 16));
      v = fmaxf(v, __shfl_xor(v, 2, 16));
      v = fmaxf(v, __shfl_xor(v, 4, 16));
      v = fmaxf(v, __shfl_xor(v, 8, 16));
      const float nm = fmaxf(mi[j], v);
      const float corr = __expf(mi[j] - nm);
      mi[j] = nm;
      p0[j] = __expf(s0a[j] - nm);
      p1[j] = __expf(s1a[j] - nm);
      float rs = p0[j] + p1[j];
      rs += __shfl_xor(rs, 1, 16);
      rs += __shfl_xor(rs, 2, 16);
      rs += __shfl_xor(rs, 4, 16);
      rs += __shfl_xor(rs, 8, 16);
      li[j] = li[j] * corr + rs;
#pragma unroll
      for (int t = 0; t < 4; ++t) acc[t][j] *= corr;
    }

    // ---- P: C-layout -> LDS -> A-layout operand ----
#pragma unroll
    for (int j = 0; j < 8; ++j) {
      pb[(j + hi8) * 32 + col]      = (_Float16)p0[j];
      pb[(j + hi8) * 32 + 16 + col] = (_Float16)p1[j];
    }
    asm volatile("s_wait_dscnt 0" ::: "memory");
    v16h pa = load_a16(pb + col * 32, base);

    // ---- O += P @ V  (V tile rows packed 32 tokens wide in LDS) ----
#pragma unroll
    for (int dt = 0; dt < 4; ++dt) {
      const _Float16* vrow = &vbuf[cur][(size_t)(dt * 16 + col) * 32];
      v16h bv = load_a16(vrow, base);
      acc[dt] = wmma16(pa, bv, acc[dt]);
    }

    __syncthreads();   // all waves done with tile i before TDM reuses buffer
  }

  // ---- normalize and store to [B, S, EMBED] f16 for the output GEMM ----
#pragma unroll
  for (int j = 0; j < 8; ++j) {
    const float inv = 1.0f / li[j];
#pragma unroll
    for (int dt = 0; dt < 4; ++dt)
      attn_ws[((size_t)b * SEQ + (s0 + j + hi8)) * EMBED + h * HD + dt * 16 + col] =
          (_Float16)(acc[dt][j] * inv);
  }
}

// ---------------------------------------------------------------------------
// Kernel 3: out = attn @ Wo^T + bo, f32 output. Wave = 16 tokens x 64 cols.
// ---------------------------------------------------------------------------
__global__ void out_proj_kernel(const _Float16* __restrict__ attn_ws,
                                const float* __restrict__ Wo,
                                const float* __restrict__ bo,
                                float* __restrict__ out) {
  const int lane = threadIdx.x & 31;
  const int wave = threadIdx.x >> 5;
  const int tok0  = blockIdx.x * 64 + wave * 16;
  const int nbase = blockIdx.y * 64;
  const int base = (lane < 16) ? 0 : 8;
  const int col  = lane & 15;
  const int hi8  = (lane < 16) ? 0 : 8;

  v8f acc[4];
#pragma unroll
  for (int t = 0; t < 4; ++t) acc[t] = (v8f){};

  const _Float16* arow = attn_ws + (size_t)(tok0 + col) * EMBED;
  for (int k0 = 0; k0 < EMBED; k0 += 32) {
    v16h a = load_a16(arow + k0, base);
#pragma unroll
    for (int nt = 0; nt < 4; ++nt) {
      const float* wrow = Wo + (size_t)(nbase + nt * 16 + col) * EMBED + k0;
      v16h bw = load_a32(wrow, base);
      acc[nt] = wmma16(a, bw, acc[nt]);
    }
  }
#pragma unroll
  for (int nt = 0; nt < 4; ++nt) {
    const float bias = bo[nbase + nt * 16 + col];
#pragma unroll
    for (int j = 0; j < 8; ++j)
      out[(size_t)(tok0 + j + hi8) * EMBED + nbase + nt * 16 + col] =
          acc[nt][j] + bias;
  }
}

// ---------------------------------------------------------------------------
extern "C" void kernel_launch(void* const* d_in, const int* in_sizes, int n_in,
                              void* d_out, int out_size, void* d_ws, size_t ws_size,
                              hipStream_t stream) {
  (void)in_sizes; (void)n_in; (void)out_size; (void)ws_size;
  const float* vin = (const float*)d_in[0];
  const float* kin = (const float*)d_in[1];
  const float* qin = (const float*)d_in[2];
  const float* Wv  = (const float*)d_in[3];
  const float* Wk  = (const float*)d_in[4];
  const float* Wq  = (const float*)d_in[5];
  const float* Wo  = (const float*)d_in[6];
  const float* bo  = (const float*)d_in[7];
  float* out = (float*)d_out;

  char* ws = (char*)d_ws;
  const size_t QKV = (size_t)BATCH * HEADS * SEQ * HD * sizeof(_Float16); // 8 MiB
  _Float16* q_ws  = (_Float16*)(ws);
  _Float16* k_ws  = (_Float16*)(ws + QKV);
  _Float16* vT_ws = (_Float16*)(ws + 2 * QKV);
  _Float16* a_ws  = (_Float16*)(ws + 3 * QKV);   // [B, S, EMBED] f16, 8 MiB

  const dim3 blk(128);  // 4 wave32 per block
  qkv_project_kernel<<<dim3(BATCH * HEADS, SEQ / 64), blk, 0, stream>>>(
      vin, kin, qin, Wv, Wk, Wq, q_ws, k_ws, vT_ws);
  flash_attn_kernel<<<dim3(BATCH * HEADS, SEQ / 64), blk, 0, stream>>>(
      q_ws, k_ws, vT_ws, a_ws);
  out_proj_kernel<<<dim3(BATCH * SEQ / 64, EMBED / 64), blk, 0, stream>>>(
      a_ws, Wo, bo, out);
}
